// PredictiveCodingNetwork_34643206210044
// MI455X (gfx1250) — compile-verified, matrix operand order checked by reading, not run on hardware
//
#include <hip/hip_runtime.h>

typedef __attribute__((ext_vector_type(16))) __bf16 v16bf;
typedef __attribute__((ext_vector_type(8)))  __bf16 v8bf;
typedef __attribute__((ext_vector_type(8)))  float  v8f;

#define BATCH 16384L
#define LRB   (0.1f / 16384.0f)
#define LDS_PAD 8   // pad LDS rows by 8 bf16 (16B) -> row stride % 64 banks = 4

// ---------------------------------------------------------------------------
// Fragment load, CDNA5 16-bit A/B layout:
// lane L holds row (L & 15); K-chunks {g*8 .. g*8+7} and {g*8+16 .. g*8+23},
// g = L>>4. Row-major source with leading dimension `ld` (elements).
// Works for global (-> global_load_b128) and LDS (-> ds_load_b128) pointers.
// ---------------------------------------------------------------------------
__device__ __forceinline__ v16bf load_frag(const __bf16* __restrict__ base,
                                           int ld, int lane) {
  const __bf16* p = base + (long)(lane & 15) * ld + ((lane >> 4) << 3);
  v8bf lo = *reinterpret_cast<const v8bf*>(p);
  v8bf hi = *reinterpret_cast<const v8bf*>(p + 16);
  v16bf f;
#pragma unroll
  for (int i = 0; i < 8; ++i) { f[i] = lo[i]; f[i + 8] = hi[i]; }
  return f;
}

__device__ __forceinline__ v8f wmma_bf16(v16bf a, v16bf b, v8f c) {
  return __builtin_amdgcn_wmma_f32_16x16x32_bf16(
      /*neg_a=*/false, a, /*neg_b=*/false, b,
      /*c_mod=*/(short)0, c, /*reuse_a=*/false, /*reuse_b=*/false);
}

// Cooperative stage of a 128-row x K bf16 slab into LDS (rows padded).
__device__ __forceinline__ void stage_b_slab(const __bf16* __restrict__ src,
                                             __bf16* lds, int K) {
  const int ldp = K + LDS_PAD;
  const int cpr = K >> 3;           // 16B chunks per row
  const int total = 128 * cpr;
  for (int c = threadIdx.x; c < total; c += 256) {
    int r  = c / cpr;
    int kc = (c - r * cpr) << 3;
    *reinterpret_cast<v8bf*>(&lds[r * ldp + kc]) =
        *reinterpret_cast<const v8bf*>(src + (long)r * K + kc);
  }
  __syncthreads();
}

// ---------------------------------------------------------------------------
// Forward layer: pred = tanh(z @ W^T + b); e = below - pred; d = e*(1-pred^2)
//   Ah: bf16 latents [BATCH x K] (K = "here")
//   Wh: bf16 weights [N x K]     (N = "below"; row n = column n of W^T)
// Block = 128(M) x 128(N); 8 waves as 4x2, each wave 32(M) x 64(N).
// B slab (128 x K) staged once in LDS; K loop barrier-free.
// ---------------------------------------------------------------------------
__global__ void __launch_bounds__(256)
pcn_forward(const __bf16* __restrict__ Ah, const __bf16* __restrict__ Wh,
            const float* __restrict__ bias, const float* __restrict__ below,
            __bf16* __restrict__ Dh, float* __restrict__ Ef,
            float* __restrict__ energy, int K, int N) {
  extern __shared__ __bf16 ldsB[];
  const int lane = threadIdx.x & 31;
  const int wave = threadIdx.x >> 5;
  const int nbn  = N >> 7;
  const int bm   = blockIdx.x / nbn;
  const int bn   = blockIdx.x % nbn;
  const int ldp  = K + LDS_PAD;

  stage_b_slab(Wh + (long)(bn << 7) * K, ldsB, K);

  const int m0  = (bm << 7) + ((wave & 3) << 5);   // wave rows: m0 .. m0+31
  const int nl0 = (wave >> 2) << 6;                // wave cols in slab: nl0 .. +63

  v8f acc[2][4] = {};
  for (int k = 0; k < K; k += 32) {
    v16bf a0 = load_frag(Ah + (long)m0 * K + k, K, lane);
    v16bf a1 = load_frag(Ah + (long)(m0 + 16) * K + k, K, lane);
#pragma unroll
    for (int j = 0; j < 4; ++j) {
      v16bf bf = load_frag(ldsB + (nl0 + j * 16) * ldp + k, ldp, lane);
      acc[0][j] = wmma_bf16(a0, bf, acc[0][j]);
      acc[1][j] = wmma_bf16(a1, bf, acc[1][j]);
    }
  }

  float local = 0.0f;
#pragma unroll
  for (int i = 0; i < 2; ++i)
#pragma unroll
    for (int j = 0; j < 4; ++j) {
      const int n = (bn << 7) + nl0 + j * 16 + (lane & 15);
      const float bn_ = bias[n];
#pragma unroll
      for (int v = 0; v < 8; ++v) {
        const int m = m0 + i * 16 + ((lane >> 4) << 3) + v;
        const long idx = (long)m * N + n;
        float p = tanhf(acc[i][j][v] + bn_);
        float e = below[idx] - p;
        Dh[idx] = (__bf16)(e * (1.0f - p * p));
        if (Ef) Ef[idx] = e;
        local += e * e;
      }
    }

  if (energy) {
    __shared__ float red[256];
    red[threadIdx.x] = local;
    __syncthreads();
    if (threadIdx.x == 0) {
      float s = 0.0f;
#pragma unroll 8
      for (int t = 0; t < 256; ++t) s += red[t];
      atomicAdd(energy, 0.5f * s / (float)BATCH);
    }
  }
}

// ---------------------------------------------------------------------------
// Latent update: z += (lr/B) * (d @ W) - (lr/B) * e_next
//   Dh: bf16 [BATCH x K] (K = "below"); Wt: bf16 [N x K] (N = "here").
// Same 128x128 blocking; writes f32 master + bf16 mirror.
// ---------------------------------------------------------------------------
__global__ void __launch_bounds__(256)
pcn_update(const __bf16* __restrict__ Dh, const __bf16* __restrict__ Wt,
           const float* __restrict__ Enext, float* __restrict__ Zf,
           __bf16* __restrict__ Zh, int K, int N) {
  extern __shared__ __bf16 ldsB[];
  const int lane = threadIdx.x & 31;
  const int wave = threadIdx.x >> 5;
  const int nbn  = N >> 7;
  const int bm   = blockIdx.x / nbn;
  const int bn   = blockIdx.x % nbn;
  const int ldp  = K + LDS_PAD;

  stage_b_slab(Wt + (long)(bn << 7) * K, ldsB, K);

  const int m0  = (bm << 7) + ((wave & 3) << 5);
  const int nl0 = (wave >> 2) << 6;

  v8f acc[2][4] = {};
  for (int k = 0; k < K; k += 32) {
    v16bf a0 = load_frag(Dh + (long)m0 * K + k, K, lane);
    v16bf a1 = load_frag(Dh + (long)(m0 + 16) * K + k, K, lane);
#pragma unroll
    for (int j = 0; j < 4; ++j) {
      v16bf bf = load_frag(ldsB + (nl0 + j * 16) * ldp + k, ldp, lane);
      acc[0][j] = wmma_bf16(a0, bf, acc[0][j]);
      acc[1][j] = wmma_bf16(a1, bf, acc[1][j]);
    }
  }

#pragma unroll
  for (int i = 0; i < 2; ++i)
#pragma unroll
    for (int j = 0; j < 4; ++j) {
      const int n = (bn << 7) + nl0 + j * 16 + (lane & 15);
#pragma unroll
      for (int v = 0; v < 8; ++v) {
        const int m = m0 + i * 16 + ((lane >> 4) << 3) + v;
        const long idx = (long)m * N + n;
        float z = Zf[idx] + LRB * acc[i][j][v];
        if (Enext) z -= LRB * Enext[idx];
        Zf[idx] = z;
        Zh[idx] = (__bf16)z;
      }
    }
}

// ---------------------------------------------------------------------------
// Helpers
// ---------------------------------------------------------------------------
__global__ void pcn_convert_w(const float* __restrict__ W, __bf16* __restrict__ Wh,
                              __bf16* __restrict__ Wt, int rows, int cols) {
  long i = (long)blockIdx.x * 256 + threadIdx.x;
  if (i < (long)rows * cols) {
    int r = (int)(i / cols), c = (int)(i % cols);
    __bf16 v = (__bf16)W[i];
    Wh[i] = v;
    Wt[(long)c * rows + r] = v;
  }
}

__global__ void pcn_zero_f32(float* __restrict__ p, long n) {
  long i = (long)blockIdx.x * 256 + threadIdx.x;
  if (i < n) p[i] = 0.0f;
}

__global__ void pcn_zero_bf16(__bf16* __restrict__ p, long n) {
  long i = (long)blockIdx.x * 256 + threadIdx.x;
  if (i < n) p[i] = (__bf16)0.0f;
}

__global__ void pcn_finalize(const float* __restrict__ zcat,
                             const float* __restrict__ energy,
                             float* __restrict__ out, long n) {
  long i = (long)blockIdx.x * 256 + threadIdx.x;
  if (i == 0) out[0] = *energy;
  if (i < n) out[1 + i] = zcat[i];
}

// ---------------------------------------------------------------------------
// Host orchestration
// ---------------------------------------------------------------------------
extern "C" void kernel_launch(void* const* d_in, const int* in_sizes, int n_in,
                              void* d_out, int out_size, void* d_ws, size_t ws_size,
                              hipStream_t stream) {
  const float* x  = (const float*)d_in[0];
  const float* W0 = (const float*)d_in[1];
  const float* b0 = (const float*)d_in[2];
  const float* W1 = (const float*)d_in[3];
  const float* b1 = (const float*)d_in[4];
  const float* W2 = (const float*)d_in[5];
  const float* b2 = (const float*)d_in[6];

  // Layer dims: k=0: here=512 below=1024 | k=1: 256/512 | k=2: 128/256
  const int here[3]  = {512, 256, 128};
  const int below[3] = {1024, 512, 256};

  char* base = (char*)d_ws;
  size_t off = 0;
  auto alloc = [&](size_t bytes) -> void* {
    void* p = base + off;
    off = (off + bytes + 255) & ~size_t(255);
    return p;
  };

  const long nzl = BATCH * (512 + 256 + 128);  // 14,680,064
  float*  zf  = (float*)alloc(nzl * 4);        // f32 latent masters (contiguous)
  __bf16* zh  = (__bf16*)alloc(nzl * 2);       // bf16 mirrors (contiguous)
  float*  z0f = zf;   float*  z1f = z0f + BATCH * 512;  float*  z2f = z1f + BATCH * 256;
  __bf16* z0h = zh;   __bf16* z1h = z0h + BATCH * 512;  __bf16* z2h = z1h + BATCH * 256;
  __bf16* d0h = (__bf16*)alloc(BATCH * 1024 * 2);
  __bf16* d1h = (__bf16*)alloc(BATCH * 512 * 2);
  __bf16* d2h = (__bf16*)alloc(BATCH * 256 * 2);
  float*  e1f = (float*)alloc(BATCH * 512 * 4);
  float*  e2f = (float*)alloc(BATCH * 256 * 4);
  __bf16* W0h = (__bf16*)alloc(1024L * 512 * 2);
  __bf16* W0t = (__bf16*)alloc(1024L * 512 * 2);
  __bf16* W1h = (__bf16*)alloc(512L * 256 * 2);
  __bf16* W1t = (__bf16*)alloc(512L * 256 * 2);
  __bf16* W2h = (__bf16*)alloc(256L * 128 * 2);
  __bf16* W2t = (__bf16*)alloc(256L * 128 * 2);
  float*  energy = (float*)alloc(256);

  const __bf16* Zh[3]  = {z0h, z1h, z2h};
  float*        Zf[3]  = {z0f, z1f, z2f};
  __bf16*       Zhm[3] = {z0h, z1h, z2h};
  const __bf16* Wh[3]  = {W0h, W1h, W2h};
  const __bf16* Wt[3]  = {W0t, W1t, W2t};
  const float*  Bias[3]  = {b0, b1, b2};
  const float*  Below[3] = {x, z0f, z1f};
  __bf16*       Dh[3] = {d0h, d1h, d2h};
  float*        Ef[3] = {nullptr, e1f, e2f};   // e0 never consumed as f32
  const float*  En[3] = {e1f, e2f, nullptr};   // e_{k+1} for the update

  auto gemm_grid = [](long M, int N) { return (unsigned)((M / 128) * (N / 128)); };
  auto smem_b    = [](int K) { return (size_t)128 * (K + LDS_PAD) * 2; };

  // --- init: zero latents + mirrors + energy scalar (every call; stateless) ---
  pcn_zero_f32 <<<(unsigned)((nzl + 255) / 256), 256, 0, stream>>>(zf, nzl);
  pcn_zero_bf16<<<(unsigned)((nzl + 255) / 256), 256, 0, stream>>>(zh, nzl);
  pcn_zero_f32 <<<1, 256, 0, stream>>>(energy, 1);

  // --- bf16 weight copies + transposes ---
  pcn_convert_w<<<(1024 * 512 + 255) / 256, 256, 0, stream>>>(W0, W0h, W0t, 1024, 512);
  pcn_convert_w<<<(512 * 256 + 255) / 256, 256, 0, stream>>>(W1, W1h, W1t, 512, 256);
  pcn_convert_w<<<(256 * 128 + 255) / 256, 256, 0, stream>>>(W2, W2h, W2t, 256, 128);

  // --- 20 inference steps ---
  for (int s = 0; s < 20; ++s) {
    for (int k = 0; k < 3; ++k)  // all e_k / d_k at current latents
      pcn_forward<<<gemm_grid(BATCH, below[k]), 256, smem_b(here[k]), stream>>>(
          Zh[k], Wh[k], Bias[k], Below[k], Dh[k], Ef[k], nullptr,
          here[k], below[k]);
    for (int k = 0; k < 3; ++k)  // simultaneous latent updates
      pcn_update<<<gemm_grid(BATCH, here[k]), 256, smem_b(below[k]), stream>>>(
          Dh[k], Wt[k], En[k], Zf[k], Zhm[k], below[k], here[k]);
  }

  // --- final free energy at converged latents ---
  for (int k = 0; k < 3; ++k)
    pcn_forward<<<gemm_grid(BATCH, below[k]), 256, smem_b(here[k]), stream>>>(
        Zh[k], Wh[k], Bias[k], Below[k], Dh[k], nullptr, energy,
        here[k], below[k]);

  // --- outputs: [F, z0, z1, z2] (zf regions are contiguous) ---
  pcn_finalize<<<(unsigned)((nzl + 255) / 256), 256, 0, stream>>>(
      zf, energy, (float*)d_out, nzl);
}
